// RWKV_7_50448685859315
// MI455X (gfx1250) — compile-verified, matrix operand order checked by reading, not run on hardware
//
#include <hip/hip_runtime.h>
#include <hip/hip_bf16.h>
#include <math.h>

#define Bc  4
#define Tc  1024
#define Dc  2048
#define Hc  32
#define Nc  64
#define BTc (Bc*Tc)

typedef __attribute__((ext_vector_type(16))) __bf16 v16bf;
typedef __attribute__((ext_vector_type(8)))  __bf16 v8bf;
typedef __attribute__((ext_vector_type(8)))  float  v8f;

union BFrag { v16bf v; v8bf h[2]; };

__device__ __forceinline__ float sigf(float x) { return 1.0f / (1.0f + __expf(-x)); }

// ---------------- weight convert + transpose: Wt[n*K+k] = bf16(W[k*N+n]) ----------------
__global__ void transpose_bf16_kernel(const float* __restrict__ W, __bf16* __restrict__ Wt,
                                      int K, int N) {
  size_t i = (size_t)blockIdx.x * blockDim.x + threadIdx.x;
  if (i >= (size_t)K * N) return;
  int k = (int)(i / N), n = (int)(i % N);
  Wt[(size_t)n * K + k] = (__bf16)W[i];
}

// ---------------- token shift + six mixes, f32 -> bf16 ----------------
__global__ void mix_kernel(const float* __restrict__ x, const float* __restrict__ x_prev,
                           const float* __restrict__ cr, const float* __restrict__ cw,
                           const float* __restrict__ ck, const float* __restrict__ cv,
                           const float* __restrict__ ca, const float* __restrict__ cg,
                           __bf16* __restrict__ xr, __bf16* __restrict__ xw,
                           __bf16* __restrict__ xk, __bf16* __restrict__ xv,
                           __bf16* __restrict__ xa, __bf16* __restrict__ xg) {
  size_t i = (size_t)blockIdx.x * blockDim.x + threadIdx.x;
  if (i >= (size_t)BTc * Dc) return;
  int d = (int)(i % Dc);
  size_t bt = i / Dc;
  int t = (int)(bt % Tc);
  int b = (int)(bt / Tc);
  float xc = x[i];
  float prev = (t == 0) ? x_prev[(size_t)b * Dc + d] : x[i - Dc];
  float xx = prev - xc;
  xr[i] = (__bf16)(xc + xx * cr[d]);
  xw[i] = (__bf16)(xc + xx * cw[d]);
  xk[i] = (__bf16)(xc + xx * ck[d]);
  xv[i] = (__bf16)(xc + xx * cv[d]);
  xa[i] = (__bf16)(xc + xx * ca[d]);
  xg[i] = (__bf16)(xc + xx * cg[d]);
}

// ---------------- bf16 WMMA GEMM: C[M,N] = A[M,K] @ Wt[N,K]^T ----------------
// 256 threads = 8 waves. Each wave: 32(M) x NT*16(N) via 2 A-frags x NT accumulators.
// Block M-tile = 256 rows. NT is compile-time (4 for N%64==0, 2 for N==32) so all
// accumulators are statically register-allocated (no MOVREL).
// ACT: 0=none, 1=tanh, 2=sigmoid.
template <int ACT, int NT, typename OUT_T>
__global__ __launch_bounds__(256) void wmma_gemm_kernel(const __bf16* __restrict__ A,
                                                        const __bf16* __restrict__ Wt,
                                                        OUT_T* __restrict__ C,
                                                        int M, int K, int N) {
  const int wave = threadIdx.x >> 5;             // 0..7
  const int lane = threadIdx.x & 31;
  const int m0   = blockIdx.x * 256 + wave * 32;
  const int n0   = blockIdx.y * 64;
  if (m0 >= M) return;                           // wave-uniform
  const int lm    = lane & 15;
  const int khalf = (lane >> 4) << 3;            // 0 or 8 (ISA 16-bit A/B layout)

  // base pointers (computed once; K-loop uses small immediate adds)
  const __bf16* ap0 = A + (size_t)(m0 + lm) * K + khalf;
  const __bf16* ap1 = ap0 + (size_t)16 * K;
  const __bf16* bp[NT];
#pragma unroll
  for (int nt = 0; nt < NT; ++nt)
    bp[nt] = Wt + (size_t)(n0 + nt * 16 + lm) * K + khalf;

  v8f acc0[NT], acc1[NT];
#pragma unroll
  for (int nt = 0; nt < NT; ++nt) { acc0[nt] = v8f{}; acc1[nt] = v8f{}; }

  for (int kk = 0; kk < K; kk += 32) {
    BFrag a0, a1;
    a0.h[0] = *(const v8bf*)(ap0 + kk);          // K = kk+khalf .. +7
    a0.h[1] = *(const v8bf*)(ap0 + kk + 16);     // K = kk+khalf+16 .. +23
    a1.h[0] = *(const v8bf*)(ap1 + kk);
    a1.h[1] = *(const v8bf*)(ap1 + kk + 16);
#pragma unroll
    for (int nt = 0; nt < NT; ++nt) {
      BFrag b;
      b.h[0] = *(const v8bf*)(bp[nt] + kk);
      b.h[1] = *(const v8bf*)(bp[nt] + kk + 16);
      acc0[nt] = __builtin_amdgcn_wmma_f32_16x16x32_bf16(
          false, a0.v, false, b.v, (short)0, acc0[nt], false, false);
      acc1[nt] = __builtin_amdgcn_wmma_f32_16x16x32_bf16(
          false, a1.v, false, b.v, (short)0, acc1[nt], false, false);
    }
  }

  // epilogue: C layout VGPR v -> M=v (lanes 0-15), M=v+8 (lanes 16-31); N = lane%16
  const int mlo = (lane >> 4) << 3;
#pragma unroll
  for (int nt = 0; nt < NT; ++nt) {
    const int n = n0 + nt * 16 + lm;
#pragma unroll
    for (int vg = 0; vg < 8; ++vg) {
      float v0 = acc0[nt][vg];
      float v1 = acc1[nt][vg];
      if (ACT == 1) { v0 = tanhf(v0); v1 = tanhf(v1); }
      else if (ACT == 2) { v0 = sigf(v0); v1 = sigf(v1); }
      C[(size_t)(m0 + vg + mlo) * N + n]      = (OUT_T)v0;
      C[(size_t)(m0 + 16 + vg + mlo) * N + n] = (OUT_T)v1;
    }
  }
}

// ---------------- elementwise post: w, a, kk-normalize, k/v updates ----------------
__global__ void post_kernel(float* wio, float* aio, float* kkio,
                            float* kio, float* vio,
                            const float* __restrict__ vfirst,
                            const float* __restrict__ w0, const float* __restrict__ a0,
                            const float* __restrict__ v0, const float* __restrict__ k_k,
                            const float* __restrict__ k_a) {
  const int bt = blockIdx.x;
  __shared__ float hn[Hc];
  if (threadIdx.x < Hc) hn[threadIdx.x] = 0.f;
  __syncthreads();
  const size_t base = (size_t)bt * Dc + (size_t)threadIdx.x * 8;
  const int h = threadIdx.x >> 3;
  float kkr[8];
  float loc = 0.f;
#pragma unroll
  for (int e = 0; e < 8; ++e) {
    const int d = threadIdx.x * 8 + e;
    const float kv = kio[base + e] * k_k[d];
    kkr[e] = kv;
    loc += kv * kv;
  }
  atomicAdd(&hn[h], loc);
  __syncthreads();
  const float inv = 1.0f / fmaxf(sqrtf(hn[h]), 1e-12f);
#pragma unroll
  for (int e = 0; e < 8; ++e) {
    const int d = threadIdx.x * 8 + e;
    const float kkn  = kkr[e] * inv;
    const float av   = sigf(a0[d] + aio[base + e]);
    const float wv   = __expf(-0.606531f * sigf(w0[d] + wio[base + e]));
    const float kn   = kio[base + e] * (1.0f + (av - 1.0f) * k_a[d]);
    const float vraw = vio[base + e];
    const float gate = sigf(v0[d] + kkio[base + e]);     // read vpre before overwrite
    const float vn   = vraw + (vfirst[base + e] - vraw) * gate;
    kkio[base + e] = kkn;
    aio[base + e]  = av;
    wio[base + e]  = wv;
    kio[base + e]  = kn;
    vio[base + e]  = vn;
  }
}

// ---------------- sequential WKV scan: one block per (b,h), 64 threads = 64 state rows --------
__global__ void scan_kernel(const float* __restrict__ r, const float* __restrict__ w,
                            const float* __restrict__ k, const float* __restrict__ v,
                            const float* __restrict__ kkv, const float* __restrict__ a,
                            const float* __restrict__ state_in,
                            float* __restrict__ y, float* __restrict__ state_out) {
  const int bh = blockIdx.x;
  const int b = bh / Hc, h = bh % Hc;
  const int i = threadIdx.x;                     // row of the state
  float st[Nc];
  const size_t sbase = (((size_t)b * Hc + h) * Nc + i) * Nc;
#pragma unroll
  for (int j = 0; j < Nc; ++j) st[j] = state_in[sbase + j];

  __shared__ float sr[Nc], sw[Nc], sk[Nc], sv[Nc], saa[Nc], sbb[Nc];

  for (int t = 0; t < Tc; ++t) {
    const size_t vbase = ((size_t)b * Tc + t) * Dc + (size_t)h * Nc;
    sr[i] = r[vbase + i];
    sw[i] = w[vbase + i];
    sk[i] = k[vbase + i];
    sv[i] = v[vbase + i];
    const float kkx = kkv[vbase + i];
    saa[i] = -kkx;
    sbb[i] = kkx * a[vbase + i];
    __syncthreads();

    float sa = 0.f;
#pragma unroll
    for (int j = 0; j < Nc; ++j) sa += st[j] * saa[j];
    const float vi = sv[i];
    float yv = 0.f;
#pragma unroll
    for (int j = 0; j < Nc; ++j) {
      st[j] = st[j] * sw[j] + sa * sbb[j] + vi * sk[j];
      yv += st[j] * sr[j];
    }
    y[vbase + i] = yv;
    __syncthreads();
  }
#pragma unroll
  for (int j = 0; j < Nc; ++j) state_out[sbase + j] = st[j];
}

// ---------------- groupnorm stats per (b,h) over (T,N) ----------------
__global__ void gn_stats_kernel(const float* __restrict__ y, float* __restrict__ stats) {
  const int bh = blockIdx.x;
  const int b = bh / Hc, h = bh % Hc;
  __shared__ float ssum[256], ssq[256];
  float s = 0.f, q = 0.f;
  for (int idx = threadIdx.x; idx < Tc * Nc; idx += 256) {
    const int t = idx / Nc, n = idx % Nc;
    const float vv = y[((size_t)b * Tc + t) * Dc + (size_t)h * Nc + n];
    s += vv; q += vv * vv;
  }
  ssum[threadIdx.x] = s; ssq[threadIdx.x] = q;
  __syncthreads();
  for (int off = 128; off > 0; off >>= 1) {
    if (threadIdx.x < off) {
      ssum[threadIdx.x] += ssum[threadIdx.x + off];
      ssq[threadIdx.x]  += ssq[threadIdx.x + off];
    }
    __syncthreads();
  }
  if (threadIdx.x == 0) {
    const float cnt = (float)(Tc * Nc);
    const float m = ssum[0] / cnt;
    const float var = ssq[0] / cnt - m * m;
    stats[bh * 2]     = m;
    stats[bh * 2 + 1] = rsqrtf(var + 0.00064f);
  }
}

// ---------------- normalize + bonus + gate, output bf16 for final GEMM ----------------
__global__ void finalize_kernel(const float* __restrict__ y, const float* __restrict__ r,
                                const float* __restrict__ k, const float* __restrict__ v,
                                const float* __restrict__ g, const float* __restrict__ r_k,
                                const float* __restrict__ ln_w, const float* __restrict__ ln_b,
                                const float* __restrict__ stats, __bf16* __restrict__ fin) {
  const int bt = blockIdx.x;
  const int b = bt / Tc;
  __shared__ float srk[Hc];
  if (threadIdx.x < Hc) srk[threadIdx.x] = 0.f;
  __syncthreads();
  const size_t base = (size_t)bt * Dc + (size_t)threadIdx.x * 8;
  const int h = threadIdx.x >> 3;
  float loc = 0.f;
#pragma unroll
  for (int e = 0; e < 8; ++e) {
    const int d = threadIdx.x * 8 + e;
    loc += r[base + e] * k[base + e] * r_k[d];
  }
  atomicAdd(&srk[h], loc);
  __syncthreads();
  const float mean = stats[(b * Hc + h) * 2];
  const float inv  = stats[(b * Hc + h) * 2 + 1];
  const float bon  = srk[h];
#pragma unroll
  for (int e = 0; e < 8; ++e) {
    const int d = threadIdx.x * 8 + e;
    const float yn = (y[base + e] - mean) * inv * ln_w[d] + ln_b[d];
    fin[base + e] = (__bf16)((yn + bon * v[base + e]) * g[base + e]);
  }
}

// ---------------- small outputs ----------------
__global__ void xlast_kernel(const float* __restrict__ x, float* __restrict__ xl) {
  const int i = blockIdx.x * blockDim.x + threadIdx.x;   // over B*D
  if (i >= Bc * Dc) return;
  const int b = i / Dc, d = i % Dc;
  xl[i] = x[((size_t)b * Tc + (Tc - 1)) * Dc + d];
}

__global__ void copyf_kernel(const float* __restrict__ s, float* __restrict__ d, size_t n) {
  size_t i = (size_t)blockIdx.x * blockDim.x + threadIdx.x;
  if (i < n) d[i] = s[i];
}

// ---------------- host-side launchers ----------------
template <int ACT, typename OUT_T>
static void launch_gemm(const __bf16* A, const __bf16* Wt, OUT_T* C,
                        int M, int K, int N, hipStream_t s) {
  dim3 grid(M / 256, (N + 63) / 64);
  if (N % 64 == 0)
    wmma_gemm_kernel<ACT, 4, OUT_T><<<grid, 256, 0, s>>>(A, Wt, C, M, K, N);
  else  // N == 32
    wmma_gemm_kernel<ACT, 2, OUT_T><<<grid, 256, 0, s>>>(A, Wt, C, M, K, N);
}

extern "C" void kernel_launch(void* const* d_in, const int* in_sizes, int n_in,
                              void* d_out, int out_size, void* d_ws, size_t ws_size,
                              hipStream_t stream) {
  const float* x       = (const float*)d_in[0];
  const float* x_prev  = (const float*)d_in[1];
  const float* v_first = (const float*)d_in[2];
  const float* state   = (const float*)d_in[3];
  const float* cr  = (const float*)d_in[4];
  const float* cw  = (const float*)d_in[5];
  const float* ck  = (const float*)d_in[6];
  const float* cv  = (const float*)d_in[7];
  const float* ca  = (const float*)d_in[8];
  const float* cg  = (const float*)d_in[9];
  const float* w0  = (const float*)d_in[10];
  const float* w1  = (const float*)d_in[11];
  const float* w2  = (const float*)d_in[12];
  const float* a0  = (const float*)d_in[13];
  const float* a1  = (const float*)d_in[14];
  const float* a2  = (const float*)d_in[15];
  const float* v0  = (const float*)d_in[16];
  const float* v1  = (const float*)d_in[17];
  const float* v2  = (const float*)d_in[18];
  const float* g1  = (const float*)d_in[19];
  const float* g2  = (const float*)d_in[20];
  const float* k_k = (const float*)d_in[21];
  const float* k_a = (const float*)d_in[22];
  const float* r_k = (const float*)d_in[23];
  const float* R_  = (const float*)d_in[24];
  const float* K_  = (const float*)d_in[25];
  const float* V_  = (const float*)d_in[26];
  const float* O_  = (const float*)d_in[27];
  const float* ln_w = (const float*)d_in[28];
  const float* ln_b = (const float*)d_in[29];
  // d_in[30..32]: H, N, layer_id (layer_id==1 per setup; layer>0 path hardcoded)

  float* out      = (float*)d_out;                       // [B,T,D]
  float* x_last   = out + (size_t)BTc * Dc;              // [B,D]
  float* state_o  = x_last + (size_t)Bc * Dc;            // [B,H,N,N]
  float* vfirst_o = state_o + (size_t)Bc * Hc * Nc * Nc; // [B,T,D]

  char* p = (char*)d_ws;
  auto carve = [&](size_t bytes) -> char* {
    char* r = p; p += (bytes + 255) & ~(size_t)255; return r;
  };

  const size_t NBD = (size_t)BTc * Dc;

  // bf16 mixed activations
  __bf16* xr = (__bf16*)carve(NBD * 2);
  __bf16* xw = (__bf16*)carve(NBD * 2);
  __bf16* xk = (__bf16*)carve(NBD * 2);
  __bf16* xv = (__bf16*)carve(NBD * 2);
  __bf16* xa = (__bf16*)carve(NBD * 2);
  __bf16* xg = (__bf16*)carve(NBD * 2);
  // bf16 transposed weights [N,K]
  __bf16* Rt  = (__bf16*)carve((size_t)Dc * Dc * 2);
  __bf16* Kt  = (__bf16*)carve((size_t)Dc * Dc * 2);
  __bf16* Vt  = (__bf16*)carve((size_t)Dc * Dc * 2);
  __bf16* Ot  = (__bf16*)carve((size_t)Dc * Dc * 2);
  __bf16* w1t = (__bf16*)carve((size_t)64 * Dc * 2);
  __bf16* w2t = (__bf16*)carve((size_t)Dc * 64 * 2);
  __bf16* a1t = (__bf16*)carve((size_t)64 * Dc * 2);
  __bf16* a2t = (__bf16*)carve((size_t)Dc * 64 * 2);
  __bf16* v1t = (__bf16*)carve((size_t)32 * Dc * 2);
  __bf16* v2t = (__bf16*)carve((size_t)Dc * 32 * 2);
  __bf16* g1t = (__bf16*)carve((size_t)128 * Dc * 2);
  __bf16* g2t = (__bf16*)carve((size_t)Dc * 128 * 2);
  // bf16 hidden activations
  __bf16* hw = (__bf16*)carve((size_t)BTc * 64 * 2);
  __bf16* ha = (__bf16*)carve((size_t)BTc * 64 * 2);
  __bf16* hv = (__bf16*)carve((size_t)BTc * 32 * 2);
  __bf16* hg = (__bf16*)carve((size_t)BTc * 128 * 2);
  // f32 intermediates
  float* rbuf = (float*)carve(NBD * 4);
  float* kbuf = (float*)carve(NBD * 4);
  float* vbuf = (float*)carve(NBD * 4);
  float* wpre = (float*)carve(NBD * 4);   // -> w (in place)
  float* apre = (float*)carve(NBD * 4);   // -> a (in place)
  float* vpre = (float*)carve(NBD * 4);   // -> kk (in place)
  float* gbuf = (float*)carve(NBD * 4);
  float* ybuf = (float*)carve(NBD * 4);
  float* stats = (float*)carve((size_t)Bc * Hc * 2 * 4);
  __bf16* fin = xr;                        // reuse: xr dead after R-projection

  auto T2 = [&](const float* W, __bf16* Wt, int K, int N) {
    size_t n = (size_t)K * N;
    transpose_bf16_kernel<<<(unsigned)((n + 255) / 256), 256, 0, stream>>>(W, Wt, K, N);
  };
  T2(R_, Rt, Dc, Dc);  T2(K_, Kt, Dc, Dc);  T2(V_, Vt, Dc, Dc);  T2(O_, Ot, Dc, Dc);
  T2(w1, w1t, Dc, 64); T2(w2, w2t, 64, Dc);
  T2(a1, a1t, Dc, 64); T2(a2, a2t, 64, Dc);
  T2(v1, v1t, Dc, 32); T2(v2, v2t, 32, Dc);
  T2(g1, g1t, Dc, 128); T2(g2, g2t, 128, Dc);

  mix_kernel<<<(unsigned)(NBD / 256), 256, 0, stream>>>(
      x, x_prev, cr, cw, ck, cv, ca, cg, xr, xw, xk, xv, xa, xg);

  // projections (WMMA bf16, f32 accumulate)
  launch_gemm<0, float>(xr, Rt, rbuf, BTc, Dc, Dc, stream);
  launch_gemm<0, float>(xk, Kt, kbuf, BTc, Dc, Dc, stream);
  launch_gemm<0, float>(xv, Vt, vbuf, BTc, Dc, Dc, stream);
  // small MLP stage 1 (fused activations into bf16 hidden)
  launch_gemm<1, __bf16>(xw, w1t, hw, BTc, Dc, 64, stream);   // tanh
  launch_gemm<0, __bf16>(xa, a1t, ha, BTc, Dc, 64, stream);
  launch_gemm<0, __bf16>(xv, v1t, hv, BTc, Dc, 32, stream);
  launch_gemm<2, __bf16>(xg, g1t, hg, BTc, Dc, 128, stream);  // sigmoid
  // small MLP stage 2
  launch_gemm<0, float>(hw, w2t, wpre, BTc, 64, Dc, stream);
  launch_gemm<0, float>(ha, a2t, apre, BTc, 64, Dc, stream);
  launch_gemm<0, float>(hv, v2t, vpre, BTc, 32, Dc, stream);
  launch_gemm<0, float>(hg, g2t, gbuf, BTc, 128, Dc, stream);

  post_kernel<<<BTc, 256, 0, stream>>>(wpre, apre, vpre, kbuf, vbuf,
                                       v_first, w0, a0, v0, k_k, k_a);

  scan_kernel<<<Bc * Hc, Nc, 0, stream>>>(rbuf, wpre, kbuf, vbuf, vpre, apre,
                                          state, ybuf, state_o);

  gn_stats_kernel<<<Bc * Hc, 256, 0, stream>>>(ybuf, stats);
  finalize_kernel<<<BTc, 256, 0, stream>>>(ybuf, rbuf, kbuf, vbuf, gbuf,
                                           r_k, ln_w, ln_b, stats, fin);

  launch_gemm<0, float>(fin, Ot, out, BTc, Dc, Dc, stream);

  xlast_kernel<<<(Bc * Dc + 255) / 256, 256, 0, stream>>>(x, x_last);
  copyf_kernel<<<(unsigned)((NBD + 255) / 256), 256, 0, stream>>>(v_first, vfirst_o, NBD);
}
